// VanillaRNN_20220706030444
// MI455X (gfx1250) — compile-verified, matrix operand order checked by reading, not run on hardware
//
#include <hip/hip_runtime.h>
#include <cmath>

#define T_DIM 2048
#define B_DIM 64
#define I_DIM 256
#define H_DIM 512
#define NWG   16   // persistent workgroups for the scan; each owns 32 of 512 H-columns

typedef __attribute__((ext_vector_type(16))) _Float16 v16h;
typedef __attribute__((ext_vector_type(8)))  _Float16 v8h;
typedef __attribute__((ext_vector_type(8)))  float    v8f;

static __device__ __forceinline__ v16h join16(v8h a, v8h b) {
  return __builtin_shufflevector(a, b, 0,1,2,3,4,5,6,7,8,9,10,11,12,13,14,15);
}

// 16-bit A/B fragment for v_wmma_f32_16x16x32_f16 from a row-major f16 matrix.
// Lane layout (ISA 7.12.2): lanes 0-15 hold k = kk+{0..7} (v0-3) and kk+{16..23} (v4-7),
// lanes 16-31 hold k = kk+{8..15} and kk+{24..31}. Per-lane: two contiguous b128 loads.
static __device__ __forceinline__ v16h frag_f16(const _Float16* base, int row,
                                                int stride, int kk, int half) {
  const _Float16* p = base + (size_t)row * stride + kk + half * 8;
  v8h lo = *(const v8h*)p;
  v8h hi = *(const v8h*)(p + 16);
  return join16(lo, hi);
}

// ---------------------------------------------------------------------------
// Init: convert Wh and Wi -> f16 once, zero h0 (f16 ping buffer) and barrier.
// Re-run every launch so graph replays are deterministic.
// ---------------------------------------------------------------------------
__global__ void init_kernel(const float* __restrict__ Wh, _Float16* __restrict__ Whh,
                            const float* __restrict__ Wi, _Float16* __restrict__ Wif,
                            _Float16* __restrict__ hbuf0, unsigned* __restrict__ bar) {
  const int idx = blockIdx.x * blockDim.x + threadIdx.x;
  if (idx < H_DIM * H_DIM) Whh[idx] = (_Float16)Wh[idx];
  if (idx < H_DIM * I_DIM) Wif[idx] = (_Float16)Wi[idx];
  if (idx < B_DIM * H_DIM) hbuf0[idx] = (_Float16)0.0f;
  if (idx < 64) bar[idx] = 0u;
}

// ---------------------------------------------------------------------------
// Phase 1: xproj[t,b,:] = x[t,b,:] @ Wi^T + bi + bh, written into d_out.
// One block (8 waves) per 16-row M-tile; each wave computes a 16x64 strip.
// A tile staged in LDS as f16 (stride padded +8 -> conflict-free ds_load_b128);
// B fragments are direct f16 b128 loads from the pre-converted Wi (L2-resident).
// ---------------------------------------------------------------------------
__global__ __launch_bounds__(256)
void xproj_kernel(const float* __restrict__ x, const _Float16* __restrict__ Wif,
                  const float* __restrict__ bi, const float* __restrict__ bh,
                  float* __restrict__ out) {
  __shared__ _Float16 lA[16 * (I_DIM + 8)];
  const int tid = threadIdx.x;
  const int r0  = blockIdx.x * 16;               // row tile in [T*B, I]

  // Stage x tile (16 x 256 f32) -> f16 LDS: 16 contiguous elements per thread.
  {
    const int e0  = tid * 16;
    const int row = e0 >> 8;                     // / I_DIM
    const int col = e0 & (I_DIM - 1);
    const float* src = x + (size_t)(r0 + row) * I_DIM + col;
    _Float16* dst = lA + row * (I_DIM + 8) + col;
#pragma unroll
    for (int j = 0; j < 16; ++j) dst[j] = (_Float16)src[j];
  }
  __syncthreads();

  const int wave = tid >> 5, lane = tid & 31;
  const int half = lane >> 4, lr = lane & 15;

  v8f acc[4] = {};
  int   ncol[4];
  float bias[4];
#pragma unroll
  for (int j = 0; j < 4; ++j) {
    ncol[j] = (wave * 4 + j) * 16 + lr;          // this lane's N column (C layout: N = lane&15)
    bias[j] = bi[ncol[j]] + bh[ncol[j]];
  }

#pragma unroll
  for (int kk = 0; kk < I_DIM; kk += 32) {
    // A fragment from LDS (ds_load_b128 x2, bank-conflict free via +8 pad)
    const _Float16* ap = &lA[lr * (I_DIM + 8) + kk + half * 8];
    v16h A = join16(*(const v8h*)ap, *(const v8h*)(ap + 16));
#pragma unroll
    for (int j = 0; j < 4; ++j) {
      v16h B = frag_f16(Wif, ncol[j], I_DIM, kk, half);  // pre-converted f16, 2x b128
      acc[j] = __builtin_amdgcn_wmma_f32_16x16x32_f16(
          false, A, false, B, (short)0, acc[j], false, false);
    }
  }

#pragma unroll
  for (int j = 0; j < 4; ++j) {
#pragma unroll
    for (int i = 0; i < 8; ++i) {
      const int m = half * 8 + i;                // C layout: vgpr i, lanes16-31 -> M = 8+i
      out[(size_t)(r0 + m) * H_DIM + ncol[j]] = acc[j][i] + bias[j];
    }
  }
}

// ---------------------------------------------------------------------------
// Phase 2: persistent scan. NWG workgroups, each owns 32 H-columns.
// Per step: h_new = tanh(xproj_t + h @ Wh^T), read/overwrite d_out[t] in place,
// f16 mirror of h double-buffered in ws for next step's A fragments.
// Each wave's Wh slice (16x512 f16 = 128 VGPRs) is preloaded into registers
// ONCE before the time loop — the hot loop is then 32 A-loads + 16 WMMAs.
// Device-wide sync per step via atomic counter (release/acquire) + s_sleep spin.
// ---------------------------------------------------------------------------
__global__ __launch_bounds__(256)
void rnn_scan_kernel(const _Float16* __restrict__ Whh, float* __restrict__ out,
                     float* __restrict__ hid, _Float16* __restrict__ hbuf,
                     unsigned* __restrict__ bar) {
  const int tid  = threadIdx.x;
  const int wave = tid >> 5, lane = tid & 31;
  const int half = lane >> 4, lr = lane & 15;
  const int mt = wave & 3, nt = wave >> 2;       // 4 M-tiles x 2 N-tiles = 8 waves
  const int m0 = mt * 16;
  const int n  = blockIdx.x * 32 + nt * 16 + lr; // this lane's owned H column

  // Register-resident B operand: this wave's Wh slice for the whole run.
  v16h Bfr[H_DIM / 32];
#pragma unroll
  for (int ks = 0; ks < H_DIM / 32; ++ks)
    Bfr[ks] = frag_f16(Whh, n, H_DIM, ks * 32, half);

  for (int t = 0; t < T_DIM; ++t) {
    const _Float16* hsrc = hbuf + (size_t)(t & 1) * (B_DIM * H_DIM);
    _Float16*       hdst = hbuf + (size_t)((t + 1) & 1) * (B_DIM * H_DIM);
    float* orow = out + (size_t)t * (B_DIM * H_DIM);

    // Prefetch this lane's xproj values (independent of the GEMM chain).
    float xp[8];
#pragma unroll
    for (int i = 0; i < 8; ++i)
      xp[i] = orow[(size_t)(m0 + half * 8 + i) * H_DIM + n];

    v8f acc = {};
#pragma unroll
    for (int ks = 0; ks < H_DIM / 32; ++ks) {
      v16h A = frag_f16(hsrc, m0 + lr, H_DIM, ks * 32, half);  // prev h, f16, L2-resident
      acc = __builtin_amdgcn_wmma_f32_16x16x32_f16(
          false, A, false, Bfr[ks], (short)0, acc, false, false);
    }

#pragma unroll
    for (int i = 0; i < 8; ++i) {
      const int b = m0 + half * 8 + i;
      const size_t idx = (size_t)b * H_DIM + n;
      const float v = tanhf(acc[i] + xp[i]);
      orow[idx] = v;
      hdst[idx] = (_Float16)v;
      if (t == T_DIM - 1) hid[idx] = v;
    }

    // device-wide step barrier
    __threadfence();
    __syncthreads();
    if (tid == 0) {
      __hip_atomic_fetch_add(bar, 1u, __ATOMIC_RELEASE, __HIP_MEMORY_SCOPE_AGENT);
      const unsigned target = (unsigned)(t + 1) * NWG;
      while (__hip_atomic_load(bar, __ATOMIC_ACQUIRE, __HIP_MEMORY_SCOPE_AGENT) < target)
        __builtin_amdgcn_s_sleep(1);
    }
    __syncthreads();
  }
}

// ---------------------------------------------------------------------------
extern "C" void kernel_launch(void* const* d_in, const int* in_sizes, int n_in,
                              void* d_out, int out_size, void* d_ws, size_t ws_size,
                              hipStream_t stream) {
  (void)in_sizes; (void)n_in; (void)out_size; (void)ws_size;
  const float* x  = (const float*)d_in[0];
  const float* Wi = (const float*)d_in[1];
  const float* bi = (const float*)d_in[2];
  const float* Wh = (const float*)d_in[3];
  const float* bh = (const float*)d_in[4];

  float* out = (float*)d_out;                               // [T,B,H] then [B,H]
  float* hid = out + (size_t)T_DIM * B_DIM * H_DIM;

  char* ws = (char*)d_ws;
  unsigned* bar  = (unsigned*)ws;                                        // 256 B
  _Float16* Whh  = (_Float16*)(ws + 256);                                // H*H f16 = 512 KB
  _Float16* Wif  = (_Float16*)(ws + 256 + (size_t)H_DIM * H_DIM * 2);    // H*I f16 = 256 KB
  _Float16* hbuf = (_Float16*)(ws + 256 + (size_t)H_DIM * H_DIM * 2
                                       + (size_t)H_DIM * I_DIM * 2);     // 2*B*H f16 = 128 KB

  init_kernel<<<(H_DIM * H_DIM + 255) / 256, 256, 0, stream>>>(Wh, Whh, Wi, Wif, hbuf, bar);
  xproj_kernel<<<(T_DIM * B_DIM) / 16, 256, 0, stream>>>(x, Wif, bi, bh, out);
  rnn_scan_kernel<<<NWG, 256, 0, stream>>>(Whh, out, hid, hbuf, bar);
}
